// GroupedQueryAttention_84524956385896
// MI455X (gfx1250) — matrix operand order repaired and checked by reading, not compiled, not run on hardware
//
#include <hip/hip_runtime.h>
#include <hip/hip_bf16.h>

// ---------------------------------------------------------------------------
// GQA attention block for MI455X (gfx1250, wave32, WMMA, bf16 data paths,
// async global->LDS double-buffered GEMM mainloop).
// ---------------------------------------------------------------------------

typedef __attribute__((ext_vector_type(16))) __bf16 v16bf;
typedef __attribute__((ext_vector_type(8)))  __bf16 v8bf;
typedef __attribute__((ext_vector_type(8)))  float  v8f;
typedef __attribute__((ext_vector_type(4)))  int    v4i;

#define T_SEQ  1024
#define DIM_   2048
#define HQ_    32
#define HKV_   8
#define HD_    64

#if defined(__AMDGCN__) &&                                              \
    __has_builtin(__builtin_amdgcn_global_load_async_to_lds_b128) &&    \
    __has_builtin(__builtin_amdgcn_s_wait_asynccnt)
#define HAVE_ASYNC 1
#else
#define HAVE_ASYNC 0
#endif

#define GAS __attribute__((address_space(1)))
#define LAS __attribute__((address_space(3)))

__device__ __forceinline__ v8f wmma_bf16(v16bf a, v16bf b, v8f c) {
  return __builtin_amdgcn_wmma_f32_16x16x32_bf16(false, a, false, b, (short)0, c,
                                                 false, false);
}

__device__ __forceinline__ v16bf cat8(v8bf a, v8bf b) {
  return __builtin_shufflevector(a, b, 0, 1, 2, 3, 4, 5, 6, 7, 8, 9, 10, 11, 12,
                                 13, 14, 15);
}

// 16-byte global -> LDS copy: async on gfx1250, sync fallback elsewhere.
__device__ __forceinline__ void async_cp16(const __bf16* g, __bf16* l) {
#if HAVE_ASYNC
  __builtin_amdgcn_global_load_async_to_lds_b128((GAS v4i*)g, (LAS v4i*)l, 0, 0);
#else
  *(v8bf*)l = *(const v8bf*)g;
#endif
}

__device__ __forceinline__ void wait_async_6() {
#if HAVE_ASYNC
  __builtin_amdgcn_s_wait_asynccnt(6);
#endif
}
__device__ __forceinline__ void wait_async_0() {
#if HAVE_ASYNC
  __builtin_amdgcn_s_wait_asynccnt(0);
#endif
}

// A fragment (16x32 bf16): lane(lo,hi): row = lo,
// elems 0..15 -> K = hi*16+j.
__device__ __forceinline__ v16bf load_a_bf(const __bf16* __restrict__ p,
                                           int row0, long ld, int k0, int lo,
                                           int hi) {
  const __bf16* r = p + (long)(row0 + lo) * ld + k0;
  v8bf x0 = *(const v8bf*)(r + hi * 16);
  v8bf x1 = *(const v8bf*)(r + hi * 16 + 8);
  return cat8(x0, x1);
}

// ---------------------------------------------------------------------------
// fp32 -> bf16 bulk convert (n multiple of 8)
// ---------------------------------------------------------------------------
__global__ __launch_bounds__(256) void cvt_f32_bf16(const float* __restrict__ x,
                                                    __bf16* __restrict__ y,
                                                    long n) {
  long i = ((long)blockIdx.x * blockDim.x + threadIdx.x) * 8;
  if (i >= n) return;
#pragma unroll
  for (int j = 0; j < 8; ++j) y[i + j] = (__bf16)x[i + j];
}

// ---------------------------------------------------------------------------
// C[M,N] = A[M,K] * W[N,K]^T, bf16 operands, f32 accumulate.
// MODE 0: store f32 row-major  MODE 1: bf16 row-major  MODE 2: bf16 C^T[n*M+m]
// 128 threads = 4 waves (2x2); wave tile 32x64; block tile 64x128.
// Mainloop: async global->LDS double buffering (ASYNCcnt), ds_load fragments.
// ---------------------------------------------------------------------------
template <int MODE>
__global__ __launch_bounds__(128) void gemm_bf16(
    const __bf16* __restrict__ A, const __bf16* __restrict__ W,
    float* __restrict__ Cf, __bf16* __restrict__ Ch, int M, int N, int K) {
  constexpr int AP = 40;  // padded LDS row stride in halves (80B: no conflicts)
  __shared__ __attribute__((aligned(16))) __bf16 As[2][64 * AP];
  __shared__ __attribute__((aligned(16))) __bf16 Ws[2][128 * AP];

  int tid  = threadIdx.x;
  int wave = tid >> 5, lane = tid & 31;
  int lo = lane & 15, hi = lane >> 4;
  int wr = wave >> 1, wc = wave & 1;
  long rowB = (long)blockIdx.y * 64;
  long colB = (long)blockIdx.x * 128;

  // stage K-slab [k0, k0+32) of both tiles into buffer `buf` (6 x b128/thread)
  auto stage = [&](int buf, int k0) {
#pragma unroll
    for (int c = 0; c < 2; ++c) {
      int chunk = tid + c * 128;  // 0..255 : A tile 64 rows x 4 chunks
      int row = chunk >> 2;
      int kk = (chunk & 3) * 8;
      async_cp16(A + (rowB + row) * K + k0 + kk, &As[buf][row * AP + kk]);
    }
#pragma unroll
    for (int c = 0; c < 4; ++c) {
      int chunk = tid + c * 128;  // 0..511 : W tile 128 rows x 4 chunks
      int row = chunk >> 2;
      int kk = (chunk & 3) * 8;
      async_cp16(W + (colB + row) * K + k0 + kk, &Ws[buf][row * AP + kk]);
    }
  };

  v8f acc[2][4];
#pragma unroll
  for (int i = 0; i < 2; ++i)
#pragma unroll
    for (int c = 0; c < 4; ++c) acc[i][c] = (v8f){};

  stage(0, 0);
  int buf = 0;
  for (int k0 = 0; k0 < K; k0 += 32) {
    bool more = (k0 + 32 < K);
    if (more) {
      stage(buf ^ 1, k0 + 32);
      wait_async_6();   // retire current slab; next slab stays in flight
    } else {
      wait_async_0();
    }
    __syncthreads();    // all waves' slab loads complete

    // A fragments for this wave's two row tiles (ds_load_b128 pairs)
    const __bf16* r0 = &As[buf][(wr * 32 + lo) * AP];
    const __bf16* r1 = &As[buf][(wr * 32 + 16 + lo) * AP];
    v16bf a0 = cat8(*(const v8bf*)(r0 + hi * 16),
                    *(const v8bf*)(r0 + hi * 16 + 8));
    v16bf a1 = cat8(*(const v8bf*)(r1 + hi * 16),
                    *(const v8bf*)(r1 + hi * 16 + 8));
#pragma unroll
    for (int c = 0; c < 4; ++c) {
      const __bf16* wrow = &Ws[buf][(wc * 64 + c * 16 + lo) * AP + hi * 16];
      v16bf bfr = cat8(*(const v8bf*)(wrow), *(const v8bf*)(wrow + 8));
      acc[0][c] = wmma_bf16(a0, bfr, acc[0][c]);
      acc[1][c] = wmma_bf16(a1, bfr, acc[1][c]);
    }
    __syncthreads();    // protect `buf` from being restaged next iteration
    buf ^= 1;
  }

  // C/D layout: vgpr r -> row = hi*8 + r, col = lo
#pragma unroll
  for (int i = 0; i < 2; ++i) {
#pragma unroll
    for (int c = 0; c < 4; ++c) {
#pragma unroll
      for (int r = 0; r < 8; ++r) {
        long row = rowB + wr * 32 + i * 16 + hi * 8 + r;
        long col = colB + wc * 64 + c * 16 + lo;
        float vv = acc[i][c][r];
        if constexpr (MODE == 0)      Cf[row * N + col] = vv;
        else if constexpr (MODE == 1) Ch[row * N + col] = (__bf16)vv;
        else                          Ch[col * M + row] = (__bf16)vv;
      }
    }
  }
}

// ---------------------------------------------------------------------------
// In-place RoPE on bf16 with output scale: x (B,T,H,64); pair i at elems
// (2i,2i+1), angle [t,i].  scale=0.125 folds 1/sqrt(HD) into Q (exact: 2^-3).
// ---------------------------------------------------------------------------
__global__ __launch_bounds__(256) void rope_bf16(__bf16* __restrict__ x,
                                                 const float* __restrict__ cs,
                                                 const float* __restrict__ sn,
                                                 int n_heads, long n_pairs,
                                                 float scale) {
  long idx = (long)blockIdx.x * blockDim.x + threadIdx.x;
  if (idx >= n_pairs) return;
  int i = (int)(idx & 31);
  long base = idx >> 5;  // (b*T + t)*H + h
  int t = (int)((base / n_heads) % T_SEQ);
  float c = cs[t * 32 + i];
  float s = sn[t * 32 + i];
  __bf16* p = x + base * 64 + 2 * i;
  float xr = (float)p[0], xi = (float)p[1];
  p[0] = (__bf16)((xr * c - xi * s) * scale);
  p[1] = (__bf16)((xr * s + xi * c) * scale);
}

// ---------------------------------------------------------------------------
// Flash-style causal GQA attention, bf16 in / bf16 out, one wave per
// (b, head, 16-row q tile); streams keys in chunks of 32 with online softmax.
// Q is pre-scaled by 1/sqrt(HD) so raw WMMA scores are final logits.
// qh: (B,T,HQ,64)  kh: (B,T,HKV,64)  vT: (HKV*64, B*T)  ob: (B,T,DIM) bf16.
// ---------------------------------------------------------------------------
__global__ __launch_bounds__(32) void attn_kernel(
    const __bf16* __restrict__ qh, const __bf16* __restrict__ kh,
    const __bf16* __restrict__ vT, __bf16* __restrict__ ob) {
  __shared__ __attribute__((aligned(16))) __bf16 pbuf[16 * 32];
  int lane = threadIdx.x & 31;
  int lo = lane & 15, hi = lane >> 4;
  int hi8 = hi * 8;
  int t0 = blockIdx.x * 16;
  int h  = blockIdx.y;
  int b  = blockIdx.z;
  int g  = h >> 2;  // kv head (N_REPS = 4)

  const __bf16* qp  = qh + ((long)b * T_SEQ * HQ_ + h) * HD_;
  const __bf16* khb = kh + (long)b * T_SEQ * (HKV_ * HD_) + g * HD_;
  const __bf16* vTb = vT + (long)g * HD_ * (4L * T_SEQ) + (long)b * T_SEQ;

  v16bf qa0 = load_a_bf(qp, t0, (long)HQ_ * HD_, 0, lo, hi);
  v16bf qa1 = load_a_bf(qp, t0, (long)HQ_ * HD_, 32, lo, hi);

  v8f oacc[4] = {{}, {}, {}, {}};
  float rmax[8], rsum[8];
#pragma unroll
  for (int r = 0; r < 8; ++r) { rmax[r] = -3.0e38f; rsum[r] = 0.0f; }

  for (int s0 = 0; s0 <= t0 + 15; s0 += 32) {
    // ---- scores: two 16x16 tiles over keys [s0, s0+32) ----
    v8f sc0 = {}, sc1 = {};
    {
      int s_a = s0 + lo;      if (s_a > T_SEQ - 1) s_a = T_SEQ - 1;
      int s_b = s0 + 16 + lo; if (s_b > T_SEQ - 1) s_b = T_SEQ - 1;
      const __bf16* kr0 = khb + (long)s_a * (HKV_ * HD_);
      const __bf16* kr1 = khb + (long)s_b * (HKV_ * HD_);
      v16bf kb;
      kb = cat8(*(const v8bf*)(kr0 + hi * 16), *(const v8bf*)(kr0 + hi * 16 + 8));
      sc0 = wmma_bf16(qa0, kb, sc0);
      kb = cat8(*(const v8bf*)(kr0 + 32 + hi * 16),
                *(const v8bf*)(kr0 + 40 + hi * 16));
      sc0 = wmma_bf16(qa1, kb, sc0);
      kb = cat8(*(const v8bf*)(kr1 + hi * 16), *(const v8bf*)(kr1 + hi * 16 + 8));
      sc1 = wmma_bf16(qa0, kb, sc1);
      kb = cat8(*(const v8bf*)(kr1 + 32 + hi * 16),
                *(const v8bf*)(kr1 + 40 + hi * 16));
      sc1 = wmma_bf16(qa1, kb, sc1);
    }

    // ---- online softmax; row = hi*8+r lives in one 16-lane half ----
#pragma unroll
    for (int r = 0; r < 8; ++r) {
      int qrow = t0 + hi8 + r;
      float v0 = (s0 + lo      <= qrow) ? sc0[r] : -1.0e30f;
      float v1 = (s0 + 16 + lo <= qrow) ? sc1[r] : -1.0e30f;
      float cmax = fmaxf(v0, v1);
      cmax = fmaxf(cmax, __shfl_xor(cmax, 1));
      cmax = fmaxf(cmax, __shfl_xor(cmax, 2));
      cmax = fmaxf(cmax, __shfl_xor(cmax, 4));
      cmax = fmaxf(cmax, __shfl_xor(cmax, 8));
      float nmax  = fmaxf(rmax[r], cmax);
      float alpha = __expf(rmax[r] - nmax);
      float p0 = __expf(v0 - nmax);
      float p1 = __expf(v1 - nmax);
      float ps = p0 + p1;
      ps += __shfl_xor(ps, 1);
      ps += __shfl_xor(ps, 2);
      ps += __shfl_xor(ps, 4);
      ps += __shfl_xor(ps, 8);
      rsum[r] = rsum[r] * alpha + ps;
      rmax[r] = nmax;
      oacc[0][r] *= alpha; oacc[1][r] *= alpha;
      oacc[2][r] *= alpha; oacc[3][r] *= alpha;
      pbuf[(hi8 + r) * 32 + lo]      = (__bf16)p0;  // (row, key) in LDS
      pbuf[(hi8 + r) * 32 + 16 + lo] = (__bf16)p1;
    }
    // single-wave workgroup: LDS ops are in-order per wave (DScnt)

    // ---- P (16x32) as A-fragment straight from LDS (2x ds_load_b128) ----
    v16bf pa = cat8(*(const v8bf*)&pbuf[lo * 32 + hi * 16],
                    *(const v8bf*)&pbuf[lo * 32 + hi * 16 + 8]);

    // ---- O += P @ V ; V^T rows are token-contiguous -> vector loads ----
#pragma unroll
    for (int dt = 0; dt < 4; ++dt) {
      const __bf16* vr = vTb + (long)(dt * 16 + lo) * (4L * T_SEQ) + s0 + hi * 16;
      v16bf vb = cat8(*(const v8bf*)vr, *(const v8bf*)(vr + 8));
      oacc[dt] = wmma_bf16(pa, vb, oacc[dt]);
    }
  }

  // ---- normalize + store bf16: ob[b, t, h*64 + d] ----
  __bf16* op = ob + (long)(b * T_SEQ + t0) * DIM_ + h * HD_;
#pragma unroll
  for (int r = 0; r < 8; ++r) {
    float inv = 1.0f / rsum[r];
#pragma unroll
    for (int dt = 0; dt < 4; ++dt)
      op[(long)(hi8 + r) * DIM_ + dt * 16 + lo] = (__bf16)(oacc[dt][r] * inv);
  }
}

// ---------------------------------------------------------------------------
extern "C" void kernel_launch(void* const* d_in, const int* in_sizes, int n_in,
                              void* d_out, int out_size, void* d_ws,
                              size_t ws_size, hipStream_t stream) {
  (void)in_sizes; (void)n_in; (void)out_size; (void)ws_size;
  const float* x  = (const float*)d_in[0];
  const float* fc = (const float*)d_in[1];
  const float* fs = (const float*)d_in[2];
  // d_in[3] = mask (causality computed inline), d_in[8] = curr_pos (== 0)
  const float* wq = (const float*)d_in[4];
  const float* wk = (const float*)d_in[5];
  const float* wv = (const float*)d_in[6];
  const float* wo = (const float*)d_in[7];
  float* out = (float*)d_out;

  const long M  = 4L * T_SEQ;        // B*T = 4096
  const long KV = (long)HKV_ * HD_;  // 512

  // bf16 workspace partition (all offsets keep >=32B alignment)
  __bf16* xh  = (__bf16*)d_ws;            // M*DIM   = 8M
  __bf16* wqh = xh  + M * DIM_;           // DIM*DIM = 4M
  __bf16* wkh = wqh + (long)DIM_ * DIM_;  // KV*DIM  = 1M
  __bf16* wvh = wkh + KV * DIM_;          // KV*DIM  = 1M
  __bf16* woh = wvh + KV * DIM_;          // DIM*DIM = 4M
  __bf16* qhb = woh + (long)DIM_ * DIM_;  // M*DIM   = 8M
  __bf16* khb = qhb + M * DIM_;           // M*KV    = 2M
  __bf16* vTb = khb + M * KV;             // KV*M    = 2M (transposed)
  __bf16* ab  = vTb + KV * M;             // M*DIM   = 8M

  // ---- convert operands to bf16 once ----
  cvt_f32_bf16<<<(int)(M * DIM_ / (8 * 256)), 256, 0, stream>>>(x, xh, M * DIM_);
  cvt_f32_bf16<<<(int)((long)DIM_ * DIM_ / (8 * 256)), 256, 0, stream>>>(
      wq, wqh, (long)DIM_ * DIM_);
  cvt_f32_bf16<<<(int)(KV * DIM_ / (8 * 256)), 256, 0, stream>>>(wk, wkh,
                                                                 KV * DIM_);
  cvt_f32_bf16<<<(int)(KV * DIM_ / (8 * 256)), 256, 0, stream>>>(wv, wvh,
                                                                 KV * DIM_);
  cvt_f32_bf16<<<(int)((long)DIM_ * DIM_ / (8 * 256)), 256, 0, stream>>>(
      wo, woh, (long)DIM_ * DIM_);

  // ---- QKV projections (WMMA, async-LDS mainloop) ----
  gemm_bf16<1><<<dim3(DIM_ / 128, (int)(M / 64)), 128, 0, stream>>>(
      xh, wqh, nullptr, qhb, (int)M, DIM_, DIM_);
  gemm_bf16<1><<<dim3((int)(KV / 128), (int)(M / 64)), 128, 0, stream>>>(
      xh, wkh, nullptr, khb, (int)M, (int)KV, DIM_);
  gemm_bf16<2><<<dim3((int)(KV / 128), (int)(M / 64)), 128, 0, stream>>>(
      xh, wvh, nullptr, vTb, (int)M, (int)KV, DIM_);  // V stored transposed

  // ---- RoPE (in place, bf16); Q also gets 1/sqrt(HD)=0.125 folded in ----
  long qpairs = M * HQ_ * 32;
  long kpairs = M * HKV_ * 32;
  rope_bf16<<<(int)((qpairs + 255) / 256), 256, 0, stream>>>(qhb, fc, fs, HQ_,
                                                             qpairs, 0.125f);
  rope_bf16<<<(int)((kpairs + 255) / 256), 256, 0, stream>>>(khb, fc, fs, HKV_,
                                                             kpairs, 1.0f);

  // ---- attention (flash, causal, GQA, WMMA) ----
  attn_kernel<<<dim3(T_SEQ / 16, HQ_, 4), 32, 0, stream>>>(qhb, khb, vTb, ab);

  // ---- output projection (WMMA, fp32 out) ----
  gemm_bf16<0><<<dim3(DIM_ / 128, (int)(M / 64)), 128, 0, stream>>>(
      ab, woh, out, nullptr, (int)M, DIM_, DIM_);
}
